// RealMamba2Block_71356586656052
// MI455X (gfx1250) — matrix operand order repaired and checked by reading, not compiled, not run on hardware
//
#include <hip/hip_runtime.h>
#include <cstdint>

#define SEQ       1024
#define DMODEL    768
#define DINNER    1536
#define DSTATE    64
#define DCONV     4
#define DTRANK    48
#define XZDIM     (2 * DINNER)            // 3072
#define XPROJDIM  (DTRANK + 2 * DSTATE)   // 176
#define DTPAD     64                      // DTRANK padded to WMMA K multiple

typedef __attribute__((ext_vector_type(16))) __bf16 v16bf;
typedef __attribute__((ext_vector_type(8)))  float  v8f;
typedef __attribute__((ext_vector_type(4)))  int    v4i;

union FragBF { v16bf v; uint4 q[2]; };

// ---- gfx1250 async global->LDS copy (ASYNCcnt path), with safe fallback ----
#if defined(__has_builtin)
#if __has_builtin(__builtin_amdgcn_global_load_async_to_lds_b128) && \
    __has_builtin(__builtin_amdgcn_s_wait_asynccnt)
#define HAVE_ASYNC_LDS 1
#endif
#endif
#ifndef HAVE_ASYNC_LDS
#define HAVE_ASYNC_LDS 0
#endif

// builtin prototype (per hipcc diagnostic): (int4 AS1*, int4 AS3*, imm, imm)
typedef __attribute__((address_space(1))) v4i* gptr_b128;
typedef __attribute__((address_space(3))) v4i* lptr_b128;

__device__ __forceinline__ void cp16_g2l(const uint16_t* g, uint16_t* l) {
#if HAVE_ASYNC_LDS
    __builtin_amdgcn_global_load_async_to_lds_b128((gptr_b128)g, (lptr_b128)l, 0, 0);
#else
    *(uint4*)l = *(const uint4*)g;
#endif
}
__device__ __forceinline__ void wait_async_keep3() {
#if HAVE_ASYNC_LDS
    __builtin_amdgcn_s_wait_asynccnt(3);   // retire all but the 3 just-issued
#endif
}
__device__ __forceinline__ void wait_async_all() {
#if HAVE_ASYNC_LDS
    __builtin_amdgcn_s_wait_asynccnt(0);
#endif
}

__device__ __forceinline__ uint16_t f2bf(float f) {
    uint32_t u = __builtin_bit_cast(uint32_t, f);
    uint32_t r = u + 0x7FFFu + ((u >> 16) & 1u);   // round-to-nearest-even
    return (uint16_t)(r >> 16);
}

// ---------------------------------------------------------------------------
// Pre-pass conversions
// ---------------------------------------------------------------------------

// f32 [K,N] row-major  ->  bf16 [N,Kpad] row-major (transposed, zero-padded K)
__global__ void k_transpose_bf16(const float* __restrict__ in, uint16_t* __restrict__ out,
                                 int K, int Kpad, int N) {
    int idx = blockIdx.x * blockDim.x + threadIdx.x;
    if (idx >= N * Kpad) return;
    int n = idx / Kpad, k = idx - n * Kpad;
    out[idx] = (k < K) ? f2bf(in[k * N + n]) : (uint16_t)0;
}

__global__ void k_convert_bf16(const float* __restrict__ in, uint16_t* __restrict__ out, int count) {
    int idx = blockIdx.x * blockDim.x + threadIdx.x;
    if (idx < count) out[idx] = f2bf(in[idx]);
}

// x_dbl [SEQ,176] -> dt_r bf16 [SEQ,64] (cols 0..47, zero pad 48..63)
__global__ void k_extract_dtr(const float* __restrict__ xdbl, uint16_t* __restrict__ out) {
    int idx = blockIdx.x * blockDim.x + threadIdx.x;
    if (idx >= SEQ * DTPAD) return;
    int m = idx >> 6, k = idx & 63;
    out[idx] = (k < DTRANK) ? f2bf(xdbl[m * XPROJDIM + k]) : (uint16_t)0;
}

// ---------------------------------------------------------------------------
// Block-tiled WMMA GEMM with async global->LDS double buffering.
// C[M,N] f32 = A[M,K] bf16 (row-major) * B^T[N,K] bf16.
// Block: 256 thr = 8 waves in 4(M)x2(N); wave computes 16x64 (4 accums).
// Block tile 64x128, K step 32. Requires M%64==0, N%128==0, K%32==0.
// EPI==1: C = softplus(C + bias[n])
// ---------------------------------------------------------------------------
template <int EPI>
__global__ __launch_bounds__(256)
void k_gemm_blk(const uint16_t* __restrict__ A, const uint16_t* __restrict__ B,
                float* __restrict__ C, int M, int N, int K, int ldc,
                const float* __restrict__ bias) {
    __shared__ uint16_t As[2][64 * 32];    // [buf][row*32 + k]
    __shared__ uint16_t Bs[2][128 * 32];

    const int tid  = threadIdx.x;
    const int lane = tid & 31;
    const int half = lane >> 4;        // K-half selector per ISA frag layout
    const int r    = lane & 15;
    const int wave = tid >> 5;
    const int mw   = wave & 3;         // 0..3 : M sub-tile
    const int nw   = wave >> 2;        // 0..1 : N 64-wide half

    const int Nb = N >> 7;
    const int bm = blockIdx.x / Nb;
    const int bn = blockIdx.x - bm * Nb;

    // copy assignments: 16B per thread for A (4KB), 2x16B for B (8KB)
    const int crow = tid >> 2, cseg = (tid & 3) * 8;
    const uint16_t* gA  = A + (size_t)(bm * 64 + crow) * K + cseg;
    const uint16_t* gB0 = B + (size_t)(bn * 128 + crow) * K + cseg;
    const uint16_t* gB1 = B + (size_t)(bn * 128 + 64 + crow) * K + cseg;

    v8f acc[4] = {{}, {}, {}, {}};

    // prologue: stage k0=0 into buffer 0
    cp16_g2l(gA, &As[0][crow * 32 + cseg]);
    cp16_g2l(gB0, &Bs[0][crow * 32 + cseg]);
    cp16_g2l(gB1, &Bs[0][(64 + crow) * 32 + cseg]);

    int cur = 0;
    for (int k0 = 0; k0 < K; k0 += 32) {
        if (k0 + 32 < K) {   // stage next buffer, then retire current buffer's copies
            int nxt = cur ^ 1;
            cp16_g2l(gA + k0 + 32, &As[nxt][crow * 32 + cseg]);
            cp16_g2l(gB0 + k0 + 32, &Bs[nxt][crow * 32 + cseg]);
            cp16_g2l(gB1 + k0 + 32, &Bs[nxt][(64 + crow) * 32 + cseg]);
            wait_async_keep3();
        } else {
            wait_async_all();
        }
        __syncthreads();

        const uint16_t* aL = &As[cur][(mw * 16 + r) * 32 + half * 8];
        FragBF a;
        a.q[0] = *(const uint4*)(aL);        // k .. k+7
        a.q[1] = *(const uint4*)(aL + 16);   // k+16 .. k+23
        const uint16_t* bBase = &Bs[cur][(nw * 64 + r) * 32 + half * 8];
#pragma unroll
        for (int j = 0; j < 4; ++j) {
            const uint16_t* bL = bBase + j * 16 * 32;
            FragBF b;
            b.q[0] = *(const uint4*)(bL);
            b.q[1] = *(const uint4*)(bL + 16);
            acc[j] = __builtin_amdgcn_wmma_f32_16x16x32_bf16(
                false, a.v, false, b.v, (short)0, acc[j], false, false);
        }
        __syncthreads();   // protect buffer about to be overwritten
        cur ^= 1;
    }

    // store: lane = n + 16*(m>=8); VGPR i holds m = i + 8*half
#pragma unroll
    for (int j = 0; j < 4; ++j) {
        const int cn = bn * 128 + nw * 64 + j * 16 + r;
        float bv = 0.0f;
        if (EPI == 1) bv = bias[cn];
        float* cPtr = C + (size_t)(bm * 64 + mw * 16 + half * 8) * ldc + cn;
#pragma unroll
        for (int i = 0; i < 8; ++i) {
            float v = acc[j][i];
            if (EPI == 1) {
                v += bv;
                v = (v > 20.0f) ? v : __logf(1.0f + __expf(v));   // softplus
            }
            cPtr[(size_t)i * ldc] = v;
        }
    }
}

// ---------------------------------------------------------------------------
// Fallback wave-per-tile WMMA GEMM (any N multiple of 16) — used for N=176.
// ---------------------------------------------------------------------------
__global__ __launch_bounds__(256)
void k_gemm_wave(const uint16_t* __restrict__ A, const uint16_t* __restrict__ B,
                 float* __restrict__ C, int M, int N, int K, int ldc) {
    const int lane  = threadIdx.x & 31;
    const int half  = lane >> 4;
    const int r     = lane & 15;
    const int gwave = (int)((blockIdx.x * blockDim.x + threadIdx.x) >> 5);
    const int nwave = (int)((gridDim.x * blockDim.x) >> 5);
    const int tilesM = M >> 4, tilesN = N >> 4;

    for (int tile = gwave; tile < tilesM * tilesN; tile += nwave) {
        const int tm = tile / tilesN, tn = tile - tm * tilesN;
        const uint16_t* aPtr = A + (size_t)(tm * 16 + r) * K + half * 8;
        const uint16_t* bPtr = B + (size_t)(tn * 16 + r) * K + half * 8;
        v8f acc = {};
        for (int k0 = 0; k0 < K; k0 += 32) {
            FragBF a, b;
            a.q[0] = *(const uint4*)(aPtr + k0);
            a.q[1] = *(const uint4*)(aPtr + k0 + 16);
            b.q[0] = *(const uint4*)(bPtr + k0);
            b.q[1] = *(const uint4*)(bPtr + k0 + 16);
            acc = __builtin_amdgcn_wmma_f32_16x16x32_bf16(
                false, a.v, false, b.v, (short)0, acc, false, false);
        }
        const int cn = tn * 16 + r;
        float* cPtr = C + (size_t)(tm * 16 + half * 8) * ldc + cn;
#pragma unroll
        for (int i = 0; i < 8; ++i) cPtr[(size_t)i * ldc] = acc[i];
    }
}

// ---------------------------------------------------------------------------
// Depthwise causal conv(4) + bias + SiLU; emits f32 (scan) and bf16 (GEMM A)
// ---------------------------------------------------------------------------
__global__ void k_conv_silu(const float* __restrict__ xz, const float* __restrict__ conv_w,
                            const float* __restrict__ conv_b,
                            float* __restrict__ u_act, uint16_t* __restrict__ u_bf) {
    int idx = blockIdx.x * blockDim.x + threadIdx.x;
    if (idx >= SEQ * DINNER) return;
    int t = idx / DINNER, d = idx - t * DINNER;
    float s = conv_b[d];
#pragma unroll
    for (int j = 0; j < DCONV; ++j) {
        int tt = t - (DCONV - 1) + j;
        if (tt >= 0) s += xz[(size_t)tt * XZDIM + d] * conv_w[d * DCONV + j];
    }
    float v = s / (1.0f + __expf(-s));   // SiLU
    u_act[idx] = v;
    u_bf[idx]  = f2bf(v);
}

// ---------------------------------------------------------------------------
// Selective scan: one wave per channel d; lane handles states n=lane, lane+32.
// ---------------------------------------------------------------------------
__global__ __launch_bounds__(256)
void k_scan(const float* __restrict__ delta, const float* __restrict__ u_act,
            const float* __restrict__ xdbl, const float* __restrict__ xz,
            const float* __restrict__ A_log, const float* __restrict__ D_skip,
            uint16_t* __restrict__ y_bf) {
    const int wave = threadIdx.x >> 5;
    const int lane = threadIdx.x & 31;
    const int d = blockIdx.x * 8 + wave;
    const int n0 = lane, n1 = lane + 32;

    const float a0 = -__expf(A_log[d * DSTATE + n0]);
    const float a1 = -__expf(A_log[d * DSTATE + n1]);
    const float Dv = D_skip[d];

    float h0 = 0.0f, h1 = 0.0f;
    for (int t = 0; t < SEQ; ++t) {
        const float dlt = delta[(size_t)t * DINNER + d];
        const float ut  = u_act[(size_t)t * DINNER + d];
        const float Bt0 = xdbl[(size_t)t * XPROJDIM + DTRANK + n0];
        const float Bt1 = xdbl[(size_t)t * XPROJDIM + DTRANK + n1];
        const float Ct0 = xdbl[(size_t)t * XPROJDIM + DTRANK + DSTATE + n0];
        const float Ct1 = xdbl[(size_t)t * XPROJDIM + DTRANK + DSTATE + n1];

        h0 = h0 * __expf(dlt * a0) + ut * Bt0;
        h1 = h1 * __expf(dlt * a1) + ut * Bt1;

        float p = h0 * Ct0 + h1 * Ct1;
#pragma unroll
        for (int off = 16; off > 0; off >>= 1) p += __shfl_xor(p, off, 32);

        if (lane == 0) {
            const float zt  = xz[(size_t)t * XZDIM + DINNER + d];
            const float sil = zt / (1.0f + __expf(-zt));
            y_bf[(size_t)t * DINNER + d] = f2bf((p + Dv * ut) * sil);
        }
    }
}

// ---------------------------------------------------------------------------
// Host orchestration
// ---------------------------------------------------------------------------
extern "C" void kernel_launch(void* const* d_in, const int* in_sizes, int n_in,
                              void* d_out, int out_size, void* d_ws, size_t ws_size,
                              hipStream_t stream) {
    (void)in_sizes; (void)n_in; (void)out_size; (void)ws_size;
    const float* x      = (const float*)d_in[0];
    const float* W_in   = (const float*)d_in[1];
    const float* conv_w = (const float*)d_in[2];
    const float* conv_b = (const float*)d_in[3];
    const float* W_xp   = (const float*)d_in[4];
    const float* W_dt   = (const float*)d_in[5];
    const float* b_dt   = (const float*)d_in[6];
    const float* A_log  = (const float*)d_in[7];
    const float* D_skip = (const float*)d_in[8];
    const float* W_out  = (const float*)d_in[9];
    float* out = (float*)d_out;

    char* ws = (char*)d_ws;
    size_t off = 0;
    auto carve = [&](size_t bytes) -> char* {
        char* p = ws + off;
        off = (off + bytes + 255) & ~(size_t)255;
        return p;
    };
    uint16_t* xbf   = (uint16_t*)carve((size_t)SEQ * DMODEL * 2);
    uint16_t* WtIn  = (uint16_t*)carve((size_t)XZDIM * DMODEL * 2);
    float*    xz    = (float*)   carve((size_t)SEQ * XZDIM * 4);
    float*    uact  = (float*)   carve((size_t)SEQ * DINNER * 4);
    uint16_t* ubf   = (uint16_t*)carve((size_t)SEQ * DINNER * 2);
    uint16_t* WtXp  = (uint16_t*)carve((size_t)XPROJDIM * DINNER * 2);
    float*    xdbl  = (float*)   carve((size_t)SEQ * XPROJDIM * 4);
    uint16_t* dtr   = (uint16_t*)carve((size_t)SEQ * DTPAD * 2);
    uint16_t* WtDt  = (uint16_t*)carve((size_t)DINNER * DTPAD * 2);
    float*    delta = (float*)   carve((size_t)SEQ * DINNER * 4);
    uint16_t* ybf   = (uint16_t*)carve((size_t)SEQ * DINNER * 2);
    uint16_t* WtOut = (uint16_t*)carve((size_t)DMODEL * DINNER * 2);

    auto cdiv = [](int a, int b) { return (a + b - 1) / b; };

    // f32 -> bf16 pre-pass (weights transposed to [N,K] for contiguous B frags)
    k_convert_bf16  <<<cdiv(SEQ * DMODEL, 256), 256, 0, stream>>>(x, xbf, SEQ * DMODEL);
    k_transpose_bf16<<<cdiv(XZDIM * DMODEL, 256), 256, 0, stream>>>(W_in, WtIn, DMODEL, DMODEL, XZDIM);
    k_transpose_bf16<<<cdiv(XPROJDIM * DINNER, 256), 256, 0, stream>>>(W_xp, WtXp, DINNER, DINNER, XPROJDIM);
    k_transpose_bf16<<<cdiv(DINNER * DTPAD, 256), 256, 0, stream>>>(W_dt, WtDt, DTRANK, DTPAD, DINNER);
    k_transpose_bf16<<<cdiv(DMODEL * DINNER, 256), 256, 0, stream>>>(W_out, WtOut, DINNER, DINNER, DMODEL);

    // GEMM1: xz = x @ W_in   [1024,768]x[768,3072]  -> 16 x 24 block tiles
    k_gemm_blk<0><<<(SEQ / 64) * (XZDIM / 128), 256, 0, stream>>>(
        xbf, WtIn, xz, SEQ, XZDIM, DMODEL, XZDIM, nullptr);

    // conv + SiLU on u half
    k_conv_silu<<<cdiv(SEQ * DINNER, 256), 256, 0, stream>>>(xz, conv_w, conv_b, uact, ubf);

    // GEMM2: x_dbl = u @ W_xproj   [1024,1536]x[1536,176]  (N=176 -> wave-tile path)
    {
        int tiles = (SEQ / 16) * (XPROJDIM / 16);
        k_gemm_wave<<<cdiv(tiles, 8), 256, 0, stream>>>(
            ubf, WtXp, xdbl, SEQ, XPROJDIM, DINNER, XPROJDIM);
    }
    k_extract_dtr<<<cdiv(SEQ * DTPAD, 256), 256, 0, stream>>>(xdbl, dtr);

    // GEMM3: delta = softplus(dt_r @ W_dt + b_dt)   [1024,64]x[64,1536]
    k_gemm_blk<1><<<(SEQ / 64) * (DINNER / 128), 256, 0, stream>>>(
        dtr, WtDt, delta, SEQ, DINNER, DTPAD, DINNER, b_dt);

    // Selective scan + D-skip + SiLU(z) gate -> y (bf16)
    k_scan<<<DINNER / 8, 256, 0, stream>>>(delta, uact, xdbl, xz, A_log, D_skip, ybf);

    // GEMM4: out = y @ W_out   [1024,1536]x[1536,768]
    k_gemm_blk<0><<<(SEQ / 64) * (DMODEL / 128), 256, 0, stream>>>(
        ybf, WtOut, out, SEQ, DMODEL, DINNER, DMODEL, nullptr);
}
